// GPTQLinear_40776419508494
// MI455X (gfx1250) — matrix operand order repaired and checked by reading, not compiled, not run on hardware
//
#include <hip/hip_runtime.h>

// GPTQ 4-bit dequant + GEMM via CDNA5 WMMA (gfx1250, wave32), double-buffered
// LDS with async global->LDS staging (GLOBAL_LOAD_ASYNC_TO_LDS_B128).
// out[256,8192] = in[256,8192]_f16 @ W[8192,8192], W = scale[g,n]*(q[k,n]-(z[g,n]+1))

typedef _Float16 v8h  __attribute__((ext_vector_type(8)));
typedef _Float16 v16h __attribute__((ext_vector_type(16)));
typedef float    v8f  __attribute__((ext_vector_type(8)));

#define IN_DIM   8192
#define OUT_DIM  8192
#define TOKENS   256
#define GROUP_SZ 128
#define BLOCK_M  128
#define BLOCK_N  128
#define BLOCK_K  32
#define NSTEPS   (IN_DIM / BLOCK_K)
#define LDA      40   // padded LDS stride (halves): 80B rows -> conflict-free 16-lane reads
#define LDB      40

#if __has_builtin(__builtin_amdgcn_global_load_async_to_lds_b128) && \
    __has_builtin(__builtin_amdgcn_s_wait_asynccnt)
#define USE_ASYNC 1
#define ASYNC_WAIT() __builtin_amdgcn_s_wait_asynccnt(0)
// Builtin expects pointers to 16-byte int vectors (per hipcc diagnostic):
//   arg0: global (addrspace 1) source, arg1: LDS (addrspace 3) dest, imm offset, imm cpol
typedef int b128_t __attribute__((vector_size(16)));
#define AS1P(p) ((__attribute__((address_space(1))) b128_t*)(p))
#define AS3P(p) ((__attribute__((address_space(3))) b128_t*)(p))
#else
#define USE_ASYNC 0
#define ASYNC_WAIT()
#endif

__global__ __launch_bounds__(256)
void gptq_wmma_f16_kernel(const _Float16* __restrict__ input,
                          const int*      __restrict__ qweight,
                          const int*      __restrict__ qzeros,
                          const _Float16* __restrict__ scales,
                          const _Float16* __restrict__ bias,
                          _Float16*       __restrict__ out)
{
    __shared__ _Float16 Alds[2][BLOCK_M * LDA];   // [m][k], 2 x 10240 B
    __shared__ _Float16 Blds[2][BLOCK_N * LDB];   // [n][k] (transposed), 2 x 10240 B

    const int t    = threadIdx.x;
    const int n0   = blockIdx.x * BLOCK_N;
    const int m0   = blockIdx.y * BLOCK_M;
    const int wave = t >> 5;
    const int lane = t & 31;
    const int wm   = (wave & 3) * 32;   // wave's M offset inside block tile
    const int wn   = (wave >> 2) * 64;  // wave's N offset inside block tile

    const int mrow  = lane & 15;        // fragment row/col within 16
    const int khalf = lane >> 4;        // 0/1: K-half selector (contiguous k order used here)

    v8f acc[2][4];
#pragma unroll
    for (int i = 0; i < 2; ++i)
#pragma unroll
        for (int j = 0; j < 4; ++j)
            acc[i][j] = (v8f){0.f, 0.f, 0.f, 0.f, 0.f, 0.f, 0.f, 0.f};

    // Each thread dequantizes 2 packed int32s per K-step: 512 = 4 packed-rows x 128 cols.
    int dq_kp[2], dq_n[2];
    float dq_s[2], dq_zs[2];
#pragma unroll
    for (int r = 0; r < 2; ++r) {
        const int idx = t + r * 256;
        dq_kp[r] = idx >> 7;     // 0..3 packed row within K-tile
        dq_n[r]  = idx & 127;    // column within N-tile
        dq_s[r]  = 0.f;
        dq_zs[r] = 0.f;
    }

    // Stage tile starting at k0 into LDS buffer `buf` (A async, B dequant).
    auto stage = [&](int k0, int buf) {
        // Refresh per-group constants (group spans 4 K-steps; stage() sees k0 ascending).
        if ((k0 & (GROUP_SZ - 1)) == 0) {
            const int g = k0 >> 7;
#pragma unroll
            for (int r = 0; r < 2; ++r) {
                const int gn = n0 + dq_n[r];
                const float s = (float)scales[g * OUT_DIM + gn];
                const unsigned zq =
                    ((unsigned)qzeros[g * (OUT_DIM / 8) + (gn >> 3)] >> ((gn & 7) * 4)) & 0xFu;
                dq_s[r]  = s;
                dq_zs[r] = (float)(zq + 1u) * s;
            }
        }
        const int kp0 = k0 >> 3;

        // ---- A tile: 128x32 halves = 512 x 16B chunks, 2 per thread ----
#pragma unroll
        for (int r = 0; r < 2; ++r) {
            const int idx = t + r * 256;
            const int m = idx >> 2, c = idx & 3;
#if USE_ASYNC
            __builtin_amdgcn_global_load_async_to_lds_b128(
                AS1P(&input[(size_t)(m0 + m) * IN_DIM + k0 + c * 8]),
                AS3P(&Alds[buf][m * LDA + c * 8]), 0, 0);
#else
            *(v8h*)&Alds[buf][m * LDA + c * 8] =
                *(const v8h*)&input[(size_t)(m0 + m) * IN_DIM + k0 + c * 8];
#endif
        }

        // ---- B tile: dequant 32x128 into LDS [n][k] ----
#pragma unroll
        for (int r = 0; r < 2; ++r) {
            const unsigned q =
                (unsigned)qweight[(size_t)(kp0 + dq_kp[r]) * OUT_DIM + n0 + dq_n[r]];
            v8h wv;
#pragma unroll
            for (int j = 0; j < 8; ++j) {
                const float w = (float)((q >> (4 * j)) & 0xFu);
                wv[j] = (_Float16)fmaf(w, dq_s[r], -dq_zs[r]);
            }
            *(v8h*)&Blds[buf][dq_n[r] * LDB + dq_kp[r] * 8] = wv;
        }

        // Prefetch the K-step after next (lowers to global_prefetch_b8).
        if (k0 + BLOCK_K < IN_DIM) {
            __builtin_prefetch(
                &qweight[(size_t)(kp0 + 4 + dq_kp[0]) * OUT_DIM + n0 + dq_n[0]], 0, 1);
        }
    };

    // ---- Software pipeline: stage(t+1) overlaps compute(t) ----
    stage(0, 0);
    ASYNC_WAIT();
    __syncthreads();

    for (int ks = 0; ks < NSTEPS; ++ks) {
        const int buf = ks & 1;
        if (ks + 1 < NSTEPS)
            stage((ks + 1) * BLOCK_K, buf ^ 1);  // buf^1 reads completed at prev barrier

        // ---- Fragments & WMMA from buffer `buf` ----
        // A 16x32 f16, k order used here (not ISA table): lane l row = l&15; lanes 0-15 hold K 0..15,
        // lanes 16-31 hold K 16..31 -> two contiguous 8-half reads.
        v16h afrag[2];
#pragma unroll
        for (int i = 0; i < 2; ++i) {
            const int m = wm + i * 16 + mrow;
            union { v16h v; v8h h[2]; } u;
            u.h[0] = *(const v8h*)&Alds[buf][m * LDA + khalf * 16];
            u.h[1] = *(const v8h*)&Alds[buf][m * LDA + khalf * 16 + 8];
            afrag[i] = u.v;
        }

#pragma unroll
        for (int j = 0; j < 4; ++j) {
            // B 32x16, same k order as A: lane l col = l&15; lanes 0-15 K=0..15, lanes 16-31 K=16..31.
            // Stored column-major in LDS -> 16 contiguous halves per lane.
            const int n = wn + j * 16 + mrow;
            const v16h bfrag = *(const v16h*)&Blds[buf][n * LDB + khalf * 16];
#pragma unroll
            for (int i = 0; i < 2; ++i) {
                acc[i][j] = __builtin_amdgcn_wmma_f32_16x16x32_f16(
                    false, afrag[i], false, bfrag, (short)0, acc[i][j], false, false);
            }
        }

        ASYNC_WAIT();       // A tile for ks+1 resident in LDS
        __syncthreads();    // dequant stores for ks+1 visible; frag reads of `buf` done
    }

    // ---- Epilogue: C/D layout m = r + 8*(lane>>4), n = lane&15 ----
#pragma unroll
    for (int i = 0; i < 2; ++i) {
#pragma unroll
        for (int j = 0; j < 4; ++j) {
            const int gn = n0 + wn + j * 16 + mrow;
            const float b = (float)bias[gn];
#pragma unroll
            for (int r = 0; r < 8; ++r) {
                const int gm = m0 + wm + i * 16 + r + (khalf << 3);
                out[(size_t)gm * OUT_DIM + gn] = (_Float16)(acc[i][j][r] + b);
            }
        }
    }
}

extern "C" void kernel_launch(void* const* d_in, const int* in_sizes, int n_in,
                              void* d_out, int out_size, void* d_ws, size_t ws_size,
                              hipStream_t stream) {
    (void)in_sizes; (void)n_in; (void)out_size; (void)d_ws; (void)ws_size;
    const _Float16* input   = (const _Float16*)d_in[0];
    const int*      qweight = (const int*)d_in[1];
    const int*      qzeros  = (const int*)d_in[2];
    const _Float16* scales  = (const _Float16*)d_in[3];
    // d_in[4] = g_idx (arange/GROUP, implied by indexing), unused
    const _Float16* bias    = (const _Float16*)d_in[5];
    _Float16*       out     = (_Float16*)d_out;

    dim3 grid(OUT_DIM / BLOCK_N, TOKENS / BLOCK_M);  // 64 x 2 = 128 workgroups
    dim3 block(256);                                  // 8 waves (wave32)
    gptq_wmma_f16_kernel<<<grid, block, 0, stream>>>(input, qweight, qzeros,
                                                     scales, bias, out);
}
